// SpatialInformerLayer_7370163880224
// MI455X (gfx1250) — matrix operand order repaired and checked by reading, not compiled, not run on hardware
//
#include <hip/hip_runtime.h>
#include <cmath>

// ---------------------------------------------------------------------------
// SpatialInformer layer for MI455X (gfx1250, wave32, WMMA 16x16x32 f16).
// Dense GEMMs (QKV proj, output proj, FFN) on v_wmma_f32_16x16x32_f16 with
// weight (B) fragments hoisted across TPW row tiles per wave and A fragments
// loaded as float2 pairs (CDNA5 A layout keeps K pairs contiguous per VGPR).
// ---------------------------------------------------------------------------

typedef __attribute__((ext_vector_type(16))) _Float16 v16h;
typedef __attribute__((ext_vector_type(8)))  float    v8f;

#define DEVFN __device__ __forceinline__

namespace cfg {
constexpr int B  = 2, N = 2048, T = 12;
constexpr int BT = B * T;                 // 24
constexpr int D  = 64, H = 4, Dh = 16;
constexpr int DFF = 32;
constexpr int SK = 24, NTOP = 24;
constexpr int ROWS  = BT * N;             // 49152
constexpr int TILES = ROWS / 16;          // 3072
constexpr int TPW   = 4;                  // row tiles per wave (B-frag reuse)
constexpr float LN_EPS = 1e-5f;
constexpr float SCALE  = 0.25f;           // 1/sqrt(Dh)
}

DEVFN v8f wmma16(const v16h& a, const v16h& b, const v8f& c) {
  // v_wmma_f32_16x16x32_f16: D = A(16x32) * B(32x16) + C
  return __builtin_amdgcn_wmma_f32_16x16x32_f16(false, a, false, b,
                                                (short)0, c, false, false);
}

// ---- CDNA5 WMMA fragment loaders (per ISA 7.12.2 layouts) -----------------
// A-matrix 16x32 f16 (MxK): lanes 0-15 -> M=lane, K in {0..7, 16..23};
// lanes 16-31 -> M=lane-16, K +8.  K pairs (2j,2j+1) contiguous -> float2.
DEVFN void load_a_frag(const float* __restrict__ A, int lda, int kbase, v16h& a) {
  const int lane  = threadIdx.x & 31;
  const int m     = lane & 15;
  const int khalf = (lane >> 4) ? 8 : 0;
#pragma unroll
  for (int j = 0; j < 8; ++j) {
    const int k = ((j < 4) ? 0 : 16) + 2 * (j & 3) + khalf + kbase;
    const float2 p = *(const float2*)(A + m * lda + k);
    a[2 * j + 0] = (_Float16)p.x;
    a[2 * j + 1] = (_Float16)p.y;
  }
}

// B-matrix 32x16 f16 (KxN): lanes 0-15 -> N=lane, K in {0..7, 16..23};
// lanes 16-31 -> N=lane-16, K +8. Column walk of row-major W (stride ldw).
DEVFN void load_b_frag(const float* __restrict__ W, int ldw, int kbase, int nbase,
                       v16h& b) {
  const int lane  = threadIdx.x & 31;
  const int n     = (lane & 15) + nbase;
  const int khalf = (lane >> 4) ? 8 : 0;
#pragma unroll
  for (int e = 0; e < 16; ++e) {
    const int k = ((e < 8) ? e : e + 8) + khalf + kbase;
    b[e] = (_Float16)W[k * ldw + n];
  }
}

// C/D 16x16 f32: VGPR g: lanes 0-15 -> M=g, N=lane; lanes 16-31 -> M=8+g.
DEVFN void store_c(float* __restrict__ O, int ldo, int nbase, const v8f& c,
                   const float* __restrict__ bias, const float* __restrict__ res,
                   bool relu) {
  const int lane = threadIdx.x & 31;
  const int n    = (lane & 15) + nbase;
  const int mof  = (lane >> 4) ? 8 : 0;
#pragma unroll
  for (int g = 0; g < 8; ++g) {
    float v = c[g] + bias[n];
    if (res) v += res[(g + mof) * ldo + n];
    if (relu) v = fmaxf(v, 0.0f);
    O[(g + mof) * ldo + n] = v;
  }
}

// ---- Kernel 1: h = x*mlp_w + mlp_b ; q/k/v = h@W + b (WMMA) ---------------
// One wave per TPW row tiles. A-tiles built in-register from one x scalar per
// row (h rows are rank-1) and reused across all three weight matrices; each
// B fragment is loaded once per wave and reused by all TPW tiles.
__global__ void __launch_bounds__(32)
k_h_qkv(const float* __restrict__ x,
        const float* __restrict__ mlp_w, const float* __restrict__ mlp_b,
        const float* __restrict__ Wq, const float* __restrict__ bq,
        const float* __restrict__ Wk, const float* __restrict__ bk,
        const float* __restrict__ Wv, const float* __restrict__ bv,
        float* __restrict__ hb, float* __restrict__ qb,
        float* __restrict__ kb2, float* __restrict__ vb) {
  using namespace cfg;
  const int tile0 = blockIdx.x * TPW;
  const int lane  = threadIdx.x & 31;
  const int m     = lane & 15;
  const int khalf = (lane >> 4) ? 8 : 0;

  float xv[TPW];
#pragma unroll
  for (int ti = 0; ti < TPW; ++ti) {
    const int r  = (tile0 + ti) * 16 + m;    // global row = bt*N + n
    const int bt = r >> 11;                  // N = 2048
    const int n  = r & 2047;
    const int b  = bt / T, t = bt % T;
    xv[ti] = x[((size_t)b * N + n) * T + t];
  }

  // store h: lane-half splits the 64 cols (each row written exactly once)
#pragma unroll
  for (int ti = 0; ti < TPW; ++ti) {
    const int r   = (tile0 + ti) * 16 + m;
    const int kb0 = (lane >> 4) * 32;
    float* hr = hb + (size_t)r * D + kb0;
#pragma unroll
    for (int kk = 0; kk < 32; kk += 4) {
      float4 hv;
      hv.x = xv[ti] * mlp_w[kb0 + kk + 0] + mlp_b[kb0 + kk + 0];
      hv.y = xv[ti] * mlp_w[kb0 + kk + 1] + mlp_b[kb0 + kk + 1];
      hv.z = xv[ti] * mlp_w[kb0 + kk + 2] + mlp_b[kb0 + kk + 2];
      hv.w = xv[ti] * mlp_w[kb0 + kk + 3] + mlp_b[kb0 + kk + 3];
      *(float4*)(hr + kk) = hv;
    }
  }

  // A fragments for all tiles (K=0..31 and K=32..63), from the rank-1 form
  v16h af[TPW][2];
#pragma unroll
  for (int ti = 0; ti < TPW; ++ti) {
#pragma unroll
    for (int e = 0; e < 16; ++e) {
      const int j = e >> 1, p = e & 1;
      const int k = ((j < 4) ? 0 : 16) + 2 * (j & 3) + p + khalf;
      af[ti][0][e] = (_Float16)(xv[ti] * mlp_w[k]      + mlp_b[k]);
      af[ti][1][e] = (_Float16)(xv[ti] * mlp_w[k + 32] + mlp_b[k + 32]);
    }
  }

  const float* Ws[3] = {Wq, Wk, Wv};
  const float* Bs[3] = {bq, bk, bv};
  float*       Os[3] = {qb, kb2, vb};
#pragma unroll
  for (int w = 0; w < 3; ++w) {
#pragma unroll
    for (int nt = 0; nt < 4; ++nt) {
      v16h bf0, bf1;                      // loaded once, reused by TPW tiles
      load_b_frag(Ws[w], D, 0,  nt * 16, bf0);
      load_b_frag(Ws[w], D, 32, nt * 16, bf1);
      const int col = (lane & 15) + nt * 16;      // 0..63
      const int hh  = col >> 4, dh = col & 15;
      const int mof = (lane >> 4) ? 8 : 0;
      const float bias = Bs[w][col];
#pragma unroll
      for (int ti = 0; ti < TPW; ++ti) {
        v8f c = {};
        c = wmma16(af[ti][0], bf0, c);
        c = wmma16(af[ti][1], bf1, c);
        // scatter into [bt][head][n][dh] layout, fused bias
#pragma unroll
        for (int g = 0; g < 8; ++g) {
          const int rr = (tile0 + ti) * 16 + g + mof;
          const int bt2 = rr >> 11, n2 = rr & 2047;
          Os[w][(((size_t)bt2 * H + hh) * N + n2) * Dh + dh] = c[g] + bias;
        }
      }
    }
  }
}

// ---- generic WMMA row-tile GEMM: O = A@W + bias (+res) (+relu) ------------
// B fragments for the whole weight matrix are kept in VGPRs and reused by
// TPW row tiles per wave (NO/16 * KD/32 <= 8 fragments = 64 VGPRs).
template <int KD, int NO, bool RELU>
__global__ void __launch_bounds__(32)
gemm_rowtile(const float* __restrict__ A, const float* __restrict__ W,
             const float* __restrict__ bias, const float* __restrict__ res,
             float* __restrict__ O) {
  using cfg::TPW;
  constexpr int NKF = KD / 32, NNT = NO / 16;

  v16h bf[NNT][NKF];
#pragma unroll
  for (int nt = 0; nt < NNT; ++nt)
#pragma unroll
    for (int kk = 0; kk < NKF; ++kk)
      load_b_frag(W, NO, kk * 32, nt * 16, bf[nt][kk]);

  const int tile0 = blockIdx.x * TPW;
#pragma unroll
  for (int ti = 0; ti < TPW; ++ti) {
    const int tile  = tile0 + ti;
    const float* Ar = A + (size_t)tile * 16 * KD;
    if (ti + 1 < TPW)
      __builtin_prefetch(Ar + 16 * KD, 0, 1);   // global_prefetch_b8 (next tile)
    float*       Or = O + (size_t)tile * 16 * NO;
    const float* Rr = res ? res + (size_t)tile * 16 * NO : nullptr;

    v16h af[NKF];
#pragma unroll
    for (int kk = 0; kk < NKF; ++kk) load_a_frag(Ar, KD, kk * 32, af[kk]);

#pragma unroll
    for (int nt = 0; nt < NNT; ++nt) {
      v8f c = {};
#pragma unroll
      for (int kk = 0; kk < NKF; ++kk) c = wmma16(af[kk], bf[nt][kk], c);
      store_c(Or, NO, nt * 16, c, bias, Rr, RELU);
    }
  }
}

// ---- Kernel 2: sampled scores -> M = max_s - sum_s/N ----------------------
DEVFN float dot16(const float4* a, const float* kr) {
  float s = 0.f;
#pragma unroll
  for (int i = 0; i < 4; ++i) {
    const float4 kv = *(const float4*)(kr + 4 * i);
    s += a[i].x * kv.x + a[i].y * kv.y + a[i].z * kv.z + a[i].w * kv.w;
  }
  return s;
}

__global__ void k_msample(const float* __restrict__ qb, const float* __restrict__ kb2,
                          const int* __restrict__ idxs, float* __restrict__ Mb) {
  using namespace cfg;
  const int gid = blockIdx.x * blockDim.x + threadIdx.x;
  if (gid >= BT * H * N) return;
  const int l  = gid & 2047;
  const int bh = gid >> 11;                       // bt*H + head
  const float* qr = qb + ((size_t)bh * N + l) * Dh;
  float4 q4[4];
#pragma unroll
  for (int i = 0; i < 4; ++i) q4[i] = *(const float4*)(qr + 4 * i);
  float mx = -3.0e38f, sum = 0.f;
#pragma unroll 4
  for (int s = 0; s < SK; ++s) {
    const int ks = idxs[l * SK + s];
    const float d = dot16(q4, kb2 + ((size_t)bh * N + ks) * Dh);
    mx = fmaxf(mx, d);
    sum += d;
  }
  Mb[gid] = mx - sum * (1.0f / (float)N);
}

// ---- Kernel 3: iterative top-24 per (bt,head), deterministic --------------
__global__ void __launch_bounds__(256)
k_topk(const float* __restrict__ Mb, int* __restrict__ mtop) {
  using namespace cfg;
  __shared__ float vals[N];
  __shared__ float rmax[256];
  __shared__ int   ridx[256];
  const int bh = blockIdx.x, tid = threadIdx.x;
  for (int i = tid; i < N; i += 256) vals[i] = Mb[(size_t)bh * N + i];
  __syncthreads();
  for (int it = 0; it < NTOP; ++it) {
    float bm = -3.0e38f; int bi = N;
    for (int i = tid; i < N; i += 256) {
      const float v = vals[i];
      if (v > bm || (v == bm && i < bi)) { bm = v; bi = i; }
    }
    rmax[tid] = bm; ridx[tid] = bi;
    __syncthreads();
    for (int s = 128; s > 0; s >>= 1) {
      if (tid < s) {
        const float vo = rmax[tid + s];
        if (vo > rmax[tid] || (vo == rmax[tid] && ridx[tid + s] < ridx[tid])) {
          rmax[tid] = vo; ridx[tid] = ridx[tid + s];
        }
      }
      __syncthreads();
    }
    if (tid == 0) { mtop[bh * NTOP + it] = ridx[0]; vals[ridx[0]] = -3.0e38f; }
    __syncthreads();
  }
}

// ---- Kernel 4: vmean[bh][dh] = mean_n v ----------------------------------
__global__ void __launch_bounds__(256)
k_vmean(const float* __restrict__ vb, float* __restrict__ vmean) {
  using namespace cfg;
  __shared__ float part[256];
  const int bh = blockIdx.x, tid = threadIdx.x;
  const int dh = tid & 15, chunk = tid >> 4;       // 16 chunks of 128
  float s = 0.f;
  for (int n = chunk * 128; n < chunk * 128 + 128; ++n)
    s += vb[((size_t)bh * N + n) * Dh + dh];
  part[tid] = s;
  __syncthreads();
  for (int st = 8; st > 0; st >>= 1) {
    if (chunk < st) part[tid] += part[tid + st * 16];
    __syncthreads();
  }
  if (chunk == 0) vmean[bh * Dh + dh] = part[dh] * (1.0f / (float)N);
}

// ---- Kernel 5: one wave per (bh,u): online softmax over N, upd = attn@v ---
__global__ void __launch_bounds__(32)
k_attn(const float* __restrict__ qb, const float* __restrict__ kb2,
       const float* __restrict__ vb, const int* __restrict__ mtop,
       float* __restrict__ upd) {
  using namespace cfg;
  const int bh = blockIdx.x / NTOP, u = blockIdx.x % NTOP;
  const int lane = threadIdx.x & 31;
  const int l = mtop[bh * NTOP + u];
  const float* qr = qb + ((size_t)bh * N + l) * Dh;
  float4 q4[4];
#pragma unroll
  for (int i = 0; i < 4; ++i) q4[i] = *(const float4*)(qr + 4 * i);

  float mrun = -3.0e38f, ssum = 0.f, acc[16];
#pragma unroll
  for (int d = 0; d < 16; ++d) acc[d] = 0.f;

  for (int n = lane; n < N; n += 32) {
    const float s = dot16(q4, kb2 + ((size_t)bh * N + n) * Dh) * SCALE;
    const float mnew = fmaxf(mrun, s);
    const float corr = __expf(mrun - mnew);
    const float w    = __expf(s - mnew);
    const float* vr = vb + ((size_t)bh * N + n) * Dh;
    ssum = ssum * corr + w;
#pragma unroll
    for (int d = 0; d < 16; ++d) acc[d] = acc[d] * corr + w * vr[d];
    mrun = mnew;
  }
  // merge 32 lane-partials (log-sum-exp safe)
#pragma unroll
  for (int off = 16; off > 0; off >>= 1) {
    const float m2 = __shfl_xor(mrun, off, 32);
    const float s2 = __shfl_xor(ssum, off, 32);
    const float mn = fmaxf(mrun, m2);
    const float c1 = __expf(mrun - mn), c2 = __expf(m2 - mn);
    ssum = ssum * c1 + s2 * c2;
#pragma unroll
    for (int d = 0; d < 16; ++d)
      acc[d] = acc[d] * c1 + __shfl_xor(acc[d], off, 32) * c2;
    mrun = mn;
  }
  if (lane == 0) {
    const float inv = 1.0f / ssum;
    float* ur = upd + ((size_t)bh * NTOP + u) * Dh;
#pragma unroll
    for (int d = 0; d < 16; ++d) ur[d] = acc[d] * inv;
  }
}

// ---- Kernel 6: build ctx rows (membership scan; no atomics) ---------------
__global__ void k_ctx(const int* __restrict__ mtop, const float* __restrict__ upd,
                      const float* __restrict__ vmean, float* __restrict__ ctx) {
  using namespace cfg;
  const int r = blockIdx.x * blockDim.x + threadIdx.x;
  if (r >= ROWS) return;
  const int bt = r >> 11, n = r & 2047;
  float row[64];
#pragma unroll
  for (int hh = 0; hh < H; ++hh) {
    const int bh = bt * H + hh;
    int found = -1;
#pragma unroll
    for (int u = 0; u < NTOP; ++u)
      if (mtop[bh * NTOP + u] == n && found < 0) found = u;
    const float* src = (found >= 0) ? upd + ((size_t)bh * NTOP + found) * Dh
                                    : vmean + (size_t)bh * Dh;
#pragma unroll
    for (int d = 0; d < 16; ++d) row[hh * 16 + d] = src[d];
  }
  float* o = ctx + (size_t)r * D;
#pragma unroll
  for (int i = 0; i < 16; ++i)
    *(float4*)(o + 4 * i) = make_float4(row[4*i], row[4*i+1], row[4*i+2], row[4*i+3]);
}

// ---- LayerNorm (row of 64) ------------------------------------------------
__global__ void k_ln(const float* __restrict__ X, const float* __restrict__ g,
                     const float* __restrict__ bb, float* __restrict__ O) {
  using namespace cfg;
  const int r = blockIdx.x * blockDim.x + threadIdx.x;
  if (r >= ROWS) return;
  const float* xr = X + (size_t)r * D;
  float row[64], mean = 0.f;
#pragma unroll
  for (int d = 0; d < D; ++d) { row[d] = xr[d]; mean += row[d]; }
  mean *= (1.0f / D);
  float var = 0.f;
#pragma unroll
  for (int d = 0; d < D; ++d) { const float c = row[d] - mean; var += c * c; }
  const float rs = rsqrtf(var * (1.0f / D) + LN_EPS);
  float* o = O + (size_t)r * D;
#pragma unroll
  for (int d = 0; d < D; ++d) o[d] = (row[d] - mean) * rs * g[d] + bb[d];
}

// ---- final: LN2 + sum over D -> out[b][n][t] ------------------------------
__global__ void k_lnsum(const float* __restrict__ X, const float* __restrict__ g,
                        const float* __restrict__ bb, float* __restrict__ out) {
  using namespace cfg;
  const int r = blockIdx.x * blockDim.x + threadIdx.x;
  if (r >= ROWS) return;
  const float* xr = X + (size_t)r * D;
  float row[64], mean = 0.f;
#pragma unroll
  for (int d = 0; d < D; ++d) { row[d] = xr[d]; mean += row[d]; }
  mean *= (1.0f / D);
  float var = 0.f;
#pragma unroll
  for (int d = 0; d < D; ++d) { const float c = row[d] - mean; var += c * c; }
  const float rs = rsqrtf(var * (1.0f / D) + LN_EPS);
  float s = 0.f;
#pragma unroll
  for (int d = 0; d < D; ++d) s += (row[d] - mean) * rs * g[d] + bb[d];
  const int bt = r >> 11, n = r & 2047;
  const int b = bt / T, t = bt % T;
  out[((size_t)b * N + n) * T + t] = s;
}

// ---------------------------------------------------------------------------
extern "C" void kernel_launch(void* const* d_in, const int* in_sizes, int n_in,
                              void* d_out, int out_size, void* d_ws, size_t ws_size,
                              hipStream_t stream) {
  using namespace cfg;
  (void)in_sizes; (void)n_in; (void)out_size; (void)ws_size;
  const float* x       = (const float*)d_in[0];
  const int*   idxs    = (const int*)  d_in[1];
  const float* mlp_w   = (const float*)d_in[2];
  const float* mlp_b   = (const float*)d_in[3];
  const float* Wq      = (const float*)d_in[4];
  const float* bq      = (const float*)d_in[5];
  const float* Wk      = (const float*)d_in[6];
  const float* bk      = (const float*)d_in[7];
  const float* Wv      = (const float*)d_in[8];
  const float* bv      = (const float*)d_in[9];
  const float* Wo      = (const float*)d_in[10];
  const float* bo      = (const float*)d_in[11];
  const float* conv1_w = (const float*)d_in[12];
  const float* conv1_b = (const float*)d_in[13];
  const float* conv2_w = (const float*)d_in[14];
  const float* conv2_b = (const float*)d_in[15];
  const float* ln1_g   = (const float*)d_in[16];
  const float* ln1_b   = (const float*)d_in[17];
  const float* ln2_g   = (const float*)d_in[18];
  const float* ln2_b   = (const float*)d_in[19];
  float* outp = (float*)d_out;

  // workspace layout (floats); total ~51.3 MB
  const size_t SZ = (size_t)ROWS * D;              // 3,145,728
  float* ws   = (float*)d_ws;
  float* hb   = ws;                                // h (also preLN2 later)
  float* qb   = ws + SZ;                           // q (also ctx, then y1)
  float* kb2  = ws + 2 * SZ;                       // k (also preLN1)
  float* vb   = ws + 3 * SZ;                       // v (also h1)
  float* Mb   = ws + 4 * SZ;                       // BT*H*N
  int*   mtop = (int*)(ws + 4 * SZ + (size_t)BT * H * N);
  float* upd  = ws + 4 * SZ + (size_t)BT * H * N + BT * H * NTOP;
  float* vmean= upd + (size_t)BT * H * NTOP * Dh;

  // 1. h + QKV projections (WMMA, B-frag reuse across TPW tiles)
  k_h_qkv<<<TILES / TPW, 32, 0, stream>>>(x, mlp_w, mlp_b, Wq, bq, Wk, bk,
                                          Wv, bv, hb, qb, kb2, vb);
  // 2. sampled sparsity measure M
  k_msample<<<(BT * H * N + 255) / 256, 256, 0, stream>>>(qb, kb2, idxs, Mb);
  // 3. top-24 rows per (bt,head)
  k_topk<<<BT * H, 256, 0, stream>>>(Mb, mtop);
  // 4. v mean over N
  k_vmean<<<BT * H, 256, 0, stream>>>(vb, vmean);
  // 5. attention for selected rows (online softmax)
  k_attn<<<BT * H * NTOP, 32, 0, stream>>>(qb, kb2, vb, mtop, upd);
  // 6. ctx rows -> qb (q no longer needed)
  k_ctx<<<(ROWS + 255) / 256, 256, 0, stream>>>(mtop, upd, vmean, qb);
  // 7. preLN1 = ctx@Wo + bo + h   -> kb2
  gemm_rowtile<64, 64, false><<<TILES / TPW, 32, 0, stream>>>(qb, Wo, bo, hb, kb2);
  // 8. h1 = LN1(preLN1)           -> vb
  k_ln<<<(ROWS + 255) / 256, 256, 0, stream>>>(kb2, ln1_g, ln1_b, vb);
  // 9. y1 = relu(h1@conv1 + b)    -> qb
  gemm_rowtile<64, 32, true><<<TILES / TPW, 32, 0, stream>>>(vb, conv1_w, conv1_b,
                                                             nullptr, qb);
  // 10. preLN2 = y1@conv2 + b + h1 -> hb
  gemm_rowtile<32, 64, false><<<TILES / TPW, 32, 0, stream>>>(qb, conv2_w, conv2_b,
                                                              vb, hb);
  // 11. LN2 + sum over D -> out
  k_lnsum<<<(ROWS + 255) / 256, 256, 0, stream>>>(hb, ln2_g, ln2_b, outp);
}
